// DecoderWithoutAttention_10986526343582
// MI455X (gfx1250) — compile-verified, matrix operand order checked by reading, not run on hardware
//
#include <hip/hip_runtime.h>

// ---------------------------------------------------------------------------
// LSTM image-captioning decoder forward for MI455X (gfx1250, wave32, WMMA).
// GEMMs: 16x16 tiles on v_wmma_f32_16x16x32_f16 (f32 accumulate).
// Fast path: weights pre-converted once to f16 in WMMA fragment-swizzled
// layout (lane-contiguous 32B), so inner loops are pure b128-load + wmma.
// ---------------------------------------------------------------------------

typedef __attribute__((ext_vector_type(16))) _Float16 v16h;
typedef __attribute__((ext_vector_type(8)))  float    v8f;

#define BB      128
#define ENCK    100352      // 14*14*512
#define EMB     512
#define HID     1024
#define G4      4096        // 4*HID
#define VOCAB   10000
#define TSTEPS  19
#define MAXCAP  20

__device__ __forceinline__ float sigmoidf(float x) { return 1.0f / (1.0f + __expf(-x)); }

__device__ __forceinline__ v8f wmma_f16(v16h a, v16h b, v8f c) {
  return __builtin_amdgcn_wmma_f32_16x16x32_f16(false, a, false, b, (short)0, c, false, false);
}

// Per-lane fragment chunk (ISA 7.12.2, 16-bit A/B 16x32):
// vector elems 0..7 = K[kbase..kbase+7], elems 8..15 = K[kbase+16..kbase+23],
// kbase = k0 + (lane>=16 ? 8 : 0). Lane&15 indexes the M row (A) / N col (B).
__device__ __forceinline__ v16h load_frag_f32(const float* __restrict__ row, int kbase) {
  v16h v;
#pragma unroll
  for (int j = 0; j < 8; ++j) {
    v[j]     = (_Float16)row[kbase + j];
    v[j + 8] = (_Float16)row[kbase + 16 + j];
  }
  return v;
}

// ---------------- sort + gather (1 block, 128 threads) ----------------------
__global__ void sort_kernel(const int* __restrict__ enc_caps,
                            const int* __restrict__ cap_lens,
                            int* __restrict__ caps_s, int* __restrict__ dec_len,
                            int* __restrict__ sort_ind, float* __restrict__ out_tail) {
  int i = threadIdx.x;
  int li = cap_lens[i];
  int rank = 0;
  for (int j = 0; j < BB; ++j) {             // stable descending rank
    int lj = cap_lens[j];
    if (lj > li || (lj == li && j < i)) rank++;
  }
  sort_ind[rank] = i;
  __syncthreads();
  int orig = sort_ind[i];
  int dl = cap_lens[orig] - 1;
  dec_len[i] = dl;
  for (int tc = 0; tc < MAXCAP; ++tc)
    caps_s[i * MAXCAP + tc] = enc_caps[orig * MAXCAP + tc];
  float* caps_out = out_tail;                // tuple outputs 2..4
  float* dlen_out = out_tail + BB * MAXCAP;
  float* sind_out = dlen_out + BB;
  for (int tc = 0; tc < MAXCAP; ++tc)
    caps_out[i * MAXCAP + tc] = (float)caps_s[i * MAXCAP + tc];
  dlen_out[i] = (float)dl;
  sind_out[i] = (float)orig;
}

// ---------------- zero h / c (and swizzled h16 if fast path) ----------------
__global__ void init_hc_kernel(float* __restrict__ h, float* __restrict__ c,
                               _Float16* __restrict__ hsw) {
  int idx = blockIdx.x * blockDim.x + threadIdx.x;   // 0 .. 131071
  h[idx] = 0.0f;
  c[idx] = 0.0f;
  if (hsw) hsw[idx] = (_Float16)0.0f;                // same element count
}

// ---------------- x_img = enc_sorted @ img_W.T + img_b ----------------------
// Single pass over img_W (205 MB, HBM-bound): convert-on-the-fly is fine.
__global__ void ximg_kernel(const float* __restrict__ enc, const float* __restrict__ img_W,
                            const float* __restrict__ img_b, const int* __restrict__ sort_ind,
                            float* __restrict__ x_img) {
  int tile = blockIdx.x;
  int tn = tile & 31, tm = tile >> 5;
  int lane = threadIdx.x, half = lane >> 4, id = lane & 15;
  const float* arow = enc   + (size_t)sort_ind[tm * 16 + id] * ENCK;
  const float* brow = img_W + (size_t)(tn * 16 + id) * ENCK;
  v8f acc0 = {}, acc1 = {};
  for (int k0 = 0; k0 < ENCK; k0 += 64) {
    int kb = k0 + half * 8;
    acc0 = wmma_f16(load_frag_f32(arow, kb),      load_frag_f32(brow, kb),      acc0);
    acc1 = wmma_f16(load_frag_f32(arow, kb + 32), load_frag_f32(brow, kb + 32), acc1);
  }
  float bias = img_b[tn * 16 + id];
#pragma unroll
  for (int r = 0; r < 8; ++r) {
    int mm = tm * 16 + (half ? 8 + r : r);
    x_img[(size_t)mm * EMB + tn * 16 + id] = acc0[r] + acc1[r] + bias;
  }
}

// ---------------- weight f32 -> f16 fragment-swizzle (once per call) --------
// out[((tile*nk + kb)*32 + lane)] = v16h fragment; wave writes 1KB contiguous.
__global__ void swz_w_kernel(const float* __restrict__ W, v16h* __restrict__ out, int K) {
  int nk = K >> 5;
  int tile = blockIdx.x / nk, kb = blockIdx.x % nk;
  int lane = threadIdx.x, half = lane >> 4, id = lane & 15;
  const float* row = W + (size_t)(tile * 16 + id) * K;
  out[((size_t)tile * nk + kb) * 32 + lane] = load_frag_f32(row, (kb << 5) + half * 8);
}

// ---------------- stage x_t (x_img or gathered embedding) as fragments ------
__global__ void stage_x_kernel(const float* __restrict__ x_img, const float* __restrict__ emb_W,
                               const int* __restrict__ caps_s, v16h* __restrict__ xsw, int t) {
  int tile = blockIdx.x >> 4, kb = blockIdx.x & 15;      // 8 x 16 blocks
  int lane = threadIdx.x, half = lane >> 4, id = lane & 15;
  int m = tile * 16 + id;
  const float* row = (t == 0) ? (x_img + (size_t)m * EMB)
                              : (emb_W + (size_t)caps_s[m * MAXCAP + (t - 1)] * EMB);
  xsw[((size_t)tile * 16 + kb) * 32 + lane] = load_frag_f32(row, (kb << 5) + half * 8);
}

// ---------------- LSTM pointwise update (+ masked h into swizzled f16) ------
__global__ void lstm_update_kernel(const float* __restrict__ gates,
                                   float* __restrict__ h, float* __restrict__ c,
                                   const int* __restrict__ dec_len, int t,
                                   _Float16* __restrict__ hsw) {
  int idx = blockIdx.x * blockDim.x + threadIdx.x;   // 0..131071
  int m = idx >> 10;
  int jj = idx & (HID - 1);
  const float* g = gates + (size_t)m * G4;
  float ig = sigmoidf(g[jj]);
  float fg = sigmoidf(g[HID + jj]);
  float gg = tanhf   (g[2 * HID + jj]);
  float og = sigmoidf(g[3 * HID + jj]);
  float cn = fg * c[idx] + ig * gg;
  float hn = og * tanhf(cn);
  bool act = dec_len[m] >= t;
  float hv = act ? hn : h[idx];
  float cv = act ? cn : c[idx];
  h[idx] = hv;
  c[idx] = cv;
  if (hsw) {
    int tm = m >> 4, id = m & 15, kb = jj >> 5, kr = jj & 31;
    int half = (kr >> 3) & 1;
    int e = (kr & 7) + ((kr & 16) ? 8 : 0);
    hsw[(((size_t)tm * 32 + kb) * 32 + half * 16 + id) * 16 + e] = (_Float16)hv;
  }
}

// ---------------- FAST gates: pure b128 loads + wmma ------------------------
__global__ void gates_fast_kernel(const v16h* __restrict__ xsw,  const v16h* __restrict__ wihsw,
                                  const v16h* __restrict__ hsw,  const v16h* __restrict__ whhsw,
                                  const float* __restrict__ b_ih, const float* __restrict__ b_hh,
                                  float* __restrict__ gates) {
  int tile = blockIdx.x;                     // tm*256 + tj
  int tj = tile & 255, tm = tile >> 8;
  int lane = threadIdx.x, half = lane >> 4, id = lane & 15;
  const v16h* ap = xsw   + (size_t)tm * 16 * 32 + lane;
  const v16h* bp = wihsw + (size_t)tj * 16 * 32 + lane;
  v8f acc0 = {}, acc1 = {};
#pragma unroll 4
  for (int kb = 0; kb < 16; kb += 2) {
    acc0 = wmma_f16(ap[kb * 32],       bp[kb * 32],       acc0);
    acc1 = wmma_f16(ap[(kb + 1) * 32], bp[(kb + 1) * 32], acc1);
  }
  const v16h* ap2 = hsw   + (size_t)tm * 32 * 32 + lane;
  const v16h* bp2 = whhsw + (size_t)tj * 32 * 32 + lane;
#pragma unroll 4
  for (int kb = 0; kb < 32; kb += 2) {
    acc0 = wmma_f16(ap2[kb * 32],       bp2[kb * 32],       acc0);
    acc1 = wmma_f16(ap2[(kb + 1) * 32], bp2[(kb + 1) * 32], acc1);
  }
  int j = tj * 16 + id;
  float bias = b_ih[j] + b_hh[j];
#pragma unroll
  for (int r = 0; r < 8; ++r) {
    int mm = tm * 16 + (half ? 8 + r : r);
    gates[(size_t)mm * G4 + j] = acc0[r] + acc1[r] + bias;
  }
}

// ---------------- FAST fc head ----------------------------------------------
__global__ void fc_fast_kernel(const v16h* __restrict__ hsw, const v16h* __restrict__ fcsw,
                               const float* __restrict__ fc_b, const int* __restrict__ dec_len,
                               float* __restrict__ preds, int t) {
  int tile = blockIdx.x;                     // tm*625 + tn
  int tn = tile % 625, tm = tile / 625;
  int lane = threadIdx.x, half = lane >> 4, id = lane & 15;
  const v16h* ap = hsw  + (size_t)tm * 32 * 32 + lane;
  const v16h* bp = fcsw + (size_t)tn * 32 * 32 + lane;
  v8f acc0 = {}, acc1 = {};
#pragma unroll 4
  for (int kb = 0; kb < 32; kb += 2) {
    acc0 = wmma_f16(ap[kb * 32],       bp[kb * 32],       acc0);
    acc1 = wmma_f16(ap[(kb + 1) * 32], bp[(kb + 1) * 32], acc1);
  }
  int n = tn * 16 + id;
  float bias = fc_b[n];
#pragma unroll
  for (int r = 0; r < 8; ++r) {
    int mm = tm * 16 + (half ? 8 + r : r);
    float v = (dec_len[mm] >= t) ? (acc0[r] + acc1[r] + bias) : 0.0f;
    preds[((size_t)mm * TSTEPS + t) * VOCAB + n] = v;
  }
}

// ---------------- FALLBACK gates / fc (f32 convert-on-the-fly) --------------
__global__ void gates_kernel(const float* __restrict__ x_img, const float* __restrict__ emb_W,
                             const int* __restrict__ caps_s,
                             const float* __restrict__ W_ih, const float* __restrict__ W_hh,
                             const float* __restrict__ b_ih, const float* __restrict__ b_hh,
                             const float* __restrict__ h, float* __restrict__ gates, int t) {
  int tile = blockIdx.x;
  int tj = tile & 255, tm = tile >> 8;
  int lane = threadIdx.x, half = lane >> 4, id = lane & 15;
  int m = tm * 16 + id, j = tj * 16 + id;
  const float* xrow = (t == 0) ? (x_img + (size_t)m * EMB)
                               : (emb_W + (size_t)caps_s[m * MAXCAP + (t - 1)] * EMB);
  const float* wrow = W_ih + (size_t)j * EMB;
  v8f acc = {};
  for (int k0 = 0; k0 < EMB; k0 += 32) {
    int kb = k0 + half * 8;
    acc = wmma_f16(load_frag_f32(xrow, kb), load_frag_f32(wrow, kb), acc);
  }
  const float* hrow = h + (size_t)m * HID;
  const float* whro = W_hh + (size_t)j * HID;
  for (int k0 = 0; k0 < HID; k0 += 32) {
    int kb = k0 + half * 8;
    acc = wmma_f16(load_frag_f32(hrow, kb), load_frag_f32(whro, kb), acc);
  }
  float bias = b_ih[j] + b_hh[j];
#pragma unroll
  for (int r = 0; r < 8; ++r) {
    int mm = tm * 16 + (half ? 8 + r : r);
    gates[(size_t)mm * G4 + j] = acc[r] + bias;
  }
}

__global__ void fc_kernel(const float* __restrict__ h, const float* __restrict__ fc_W,
                          const float* __restrict__ fc_b, const int* __restrict__ dec_len,
                          float* __restrict__ preds, int t) {
  int tile = blockIdx.x;
  int tn = tile % 625, tm = tile / 625;
  int lane = threadIdx.x, half = lane >> 4, id = lane & 15;
  const float* arow = h    + (size_t)(tm * 16 + id) * HID;
  const float* brow = fc_W + (size_t)(tn * 16 + id) * HID;
  v8f acc = {};
  for (int k0 = 0; k0 < HID; k0 += 32) {
    int kb = k0 + half * 8;
    acc = wmma_f16(load_frag_f32(arow, kb), load_frag_f32(brow, kb), acc);
  }
  int n = tn * 16 + id;
  float bias = fc_b[n];
#pragma unroll
  for (int r = 0; r < 8; ++r) {
    int mm = tm * 16 + (half ? 8 + r : r);
    float v = (dec_len[mm] >= t) ? (acc[r] + bias) : 0.0f;
    preds[((size_t)mm * TSTEPS + t) * VOCAB + n] = v;
  }
}

// ---------------------------------------------------------------------------
extern "C" void kernel_launch(void* const* d_in, const int* in_sizes, int n_in,
                              void* d_out, int out_size, void* d_ws, size_t ws_size,
                              hipStream_t stream) {
  const float* encoder_out = (const float*)d_in[0];
  const int*   enc_caps    = (const int*)  d_in[1];
  const int*   cap_lens    = (const int*)  d_in[2];
  const float* emb_W       = (const float*)d_in[3];
  const float* img_W       = (const float*)d_in[4];
  const float* img_b       = (const float*)d_in[5];
  const float* W_ih        = (const float*)d_in[6];
  const float* W_hh        = (const float*)d_in[7];
  const float* b_ih        = (const float*)d_in[8];
  const float* b_hh        = (const float*)d_in[9];
  const float* fc_W        = (const float*)d_in[10];
  const float* fc_b        = (const float*)d_in[11];

  // ---- workspace layout: f32 region, int region, then 32B-aligned f16 region
  float* ws     = (float*)d_ws;
  float* x_img  = ws;                               // 128*512
  float* h      = x_img + BB * EMB;                 // 128*1024
  float* c      = h + BB * HID;                     // 128*1024
  float* gates  = c + BB * HID;                     // 128*4096
  int*   caps_s = (int*)(gates + (size_t)BB * G4);  // 128*20
  int*   dlen   = caps_s + BB * MAXCAP;             // 128
  int*   sidx   = dlen + BB;                        // 128 (+pad to 32B boundary)
  v16h*  hsw    = (v16h*)(sidx + BB + 96);          // 8*32*32 frags   (256 KB)
  v16h*  xsw    = hsw + 8 * 32 * 32;                // 8*16*32 frags   (128 KB)
  v16h*  wihsw  = xsw + 8 * 16 * 32;                // 256*16*32 frags (  4 MB)
  v16h*  whhsw  = wihsw + 256 * 16 * 32;            // 256*32*32 frags (  8 MB)
  v16h*  fcsw   = whhsw + 256 * 32 * 32;            // 625*32*32 frags ( 20 MB)
  size_t needed = (size_t)((char*)(fcsw + 625 * 32 * 32) - (char*)d_ws);
  bool   fast   = ws_size >= needed;

  float* preds  = (float*)d_out;                         // 128*19*10000
  float* tail   = preds + (size_t)BB * TSTEPS * VOCAB;   // caps, dec_len, sort_ind

  init_hc_kernel<<<(BB * HID) / 256, 256, 0, stream>>>(h, c, fast ? (_Float16*)hsw : nullptr);
  sort_kernel<<<1, BB, 0, stream>>>(enc_caps, cap_lens, caps_s, dlen, sidx, tail);
  ximg_kernel<<<(BB / 16) * (EMB / 16), 32, 0, stream>>>(encoder_out, img_W, img_b, sidx, x_img);

  if (fast) {
    swz_w_kernel<<<256 * 16, 32, 0, stream>>>(W_ih, wihsw, EMB);
    swz_w_kernel<<<256 * 32, 32, 0, stream>>>(W_hh, whhsw, HID);
    swz_w_kernel<<<625 * 32, 32, 0, stream>>>(fc_W, fcsw, HID);
  }

  for (int t = 0; t < TSTEPS; ++t) {
    if (fast) {
      stage_x_kernel<<<8 * 16, 32, 0, stream>>>(x_img, emb_W, caps_s, xsw, t);
      gates_fast_kernel<<<(BB / 16) * (G4 / 16), 32, 0, stream>>>(
          xsw, wihsw, (const v16h*)hsw, whhsw, b_ih, b_hh, gates);
      lstm_update_kernel<<<(BB * HID) / 256, 256, 0, stream>>>(
          gates, h, c, dlen, t, (_Float16*)hsw);
      fc_fast_kernel<<<(BB / 16) * (VOCAB / 16), 32, 0, stream>>>(
          (const v16h*)hsw, fcsw, fc_b, dlen, preds, t);
    } else {
      gates_kernel<<<(BB / 16) * (G4 / 16), 32, 0, stream>>>(
          x_img, emb_W, caps_s, W_ih, W_hh, b_ih, b_hh, h, gates, t);
      lstm_update_kernel<<<(BB * HID) / 256, 256, 0, stream>>>(gates, h, c, dlen, t, nullptr);
      fc_kernel<<<(BB / 16) * (VOCAB / 16), 32, 0, stream>>>(h, fc_W, fc_b, dlen, preds, t);
    }
  }
}
